// HDQN_KMeans_10668698763654
// MI455X (gfx1250) — matrix operand order repaired and checked by reading, not compiled, not run on hardware
//
#include <hip/hip_runtime.h>
#include <hip/hip_bf16.h>

typedef __attribute__((ext_vector_type(16))) __bf16 v16bf;
typedef __attribute__((ext_vector_type(8)))  float  v8f;
typedef __attribute__((ext_vector_type(4)))  unsigned int v4u;

#define DECAY     0.99f
#define EPS       1e-5f
#define NCLUST    512
#define DIM       512
#define NPOINTS   131072

// ---- main kernel tiling ----
#define WG_THREADS 128      // 4 waves of 32
#define WAVES      4
#define ROWS_PER_WAVE 16
#define ROWS_PER_WG   64    // WAVES * 16
#define LDS_STRIDE    520   // 512 + 8 pad (bf16 elems) to break bank alignment
#define GRP_CLUST     64    // centroids staged per group (4 N-tiles of 16)
#define NTILES        4
#define NGROUPS       8     // 512 / 64
#define KCHUNKS       16    // 512 / 32

__device__ __forceinline__ unsigned short f2bf(float x) {
    unsigned int u = __float_as_uint(x);
    u += 0x7FFFu + ((u >> 16) & 1u);          // round-to-nearest-even
    return (unsigned short)(u >> 16);
}

__device__ __forceinline__ void st_bf16x4(unsigned short* p, float a, float b, float c, float d) {
    uint2 q;
    q.x = (unsigned)f2bf(a) | ((unsigned)f2bf(b) << 16);
    q.y = (unsigned)f2bf(c) | ((unsigned)f2bf(d) << 16);
    *(uint2*)p = q;                           // 8B-aligned LDS store
}

union FragCvt { struct { v4u lo, hi; } u; v16bf v; };

// A fragment (16x32 bf16, ISA 7.12.2): lane half selects kb=0/8; elems [kb..kb+7] and [kb+16..kb+23]
__device__ __forceinline__ v16bf ld_frag_a(const unsigned short* p) {
    FragCvt c;
    c.u.lo = *(const v4u*)(p);
    c.u.hi = *(const v4u*)(p + 16);
    return c.v;
}
// B fragment (32x16 bf16): lane half selects kb=0/8; elems [kb..kb+7] and [kb+16..kb+23]
__device__ __forceinline__ v16bf ld_frag_b(const unsigned short* p) {
    FragCvt c;
    c.u.lo = *(const v4u*)(p);
    c.u.hi = *(const v4u*)(p + 16);
    return c.v;
}

// ---------------------------------------------------------------------------
__global__ void __launch_bounds__(WG_THREADS)
kmeans_assign_kernel(const float* __restrict__ X,
                     const float* __restrict__ centroids,
                     const float* __restrict__ c2,
                     float* __restrict__ out_q,
                     float* __restrict__ out_idx,
                     float* __restrict__ counts,
                     float* __restrict__ dw) {
    __shared__ unsigned short Alds[ROWS_PER_WG * LDS_STRIDE];  // 66,560 B
    __shared__ unsigned short Blds[GRP_CLUST  * LDS_STRIDE];   // 66,560 B

    const int tid  = threadIdx.x;
    const int lane = tid & 31;
    const int w    = tid >> 5;
    const size_t rowBase = (size_t)blockIdx.x * ROWS_PER_WG;

    // ---- Phase 1: normalize this block's 64 X rows into LDS as bf16 ----
    for (int r = 0; r < ROWS_PER_WAVE; ++r) {
        const int lrow = w * ROWS_PER_WAVE + r;
        const float* xp = X + (rowBase + lrow) * DIM;
        float4 v[4];
        float ss = 0.f;
        #pragma unroll
        for (int i = 0; i < 4; ++i) {
            v[i] = *(const float4*)(xp + i * 128 + lane * 4);
            ss += v[i].x * v[i].x + v[i].y * v[i].y + v[i].z * v[i].z + v[i].w * v[i].w;
        }
        #pragma unroll
        for (int m = 16; m >= 1; m >>= 1) ss += __shfl_xor(ss, m, 32);
        const float rn = rsqrtf(ss);
        #pragma unroll
        for (int i = 0; i < 4; ++i)
            st_bf16x4(&Alds[lrow * LDS_STRIDE + i * 128 + lane * 4],
                      v[i].x * rn, v[i].y * rn, v[i].z * rn, v[i].w * rn);
    }

    // ---- Phase 2: distances via bf16 WMMA + running argmin ----
    float best_d[8];
    int   best_i[8];
    #pragma unroll
    for (int j = 0; j < 8; ++j) { best_d[j] = 3.4e38f; best_i[j] = 0; }

    const int arow = w * ROWS_PER_WAVE + (lane & 15);
    const int kb   = (lane >> 4) * 8;          // A half-wave K offset
    const int bko  = (lane >> 4) * 8;          // B half-wave K offset

    for (int grp = 0; grp < NGROUPS; ++grp) {
        __syncthreads();                        // previous group's compute done
        const int cbase = grp * GRP_CLUST;
        // stage 64 centroid rows f32 -> bf16 LDS
        for (int q = tid; q < GRP_CLUST * (DIM / 4); q += WG_THREADS) {
            const int br = q >> 7;              // row in group
            const int c4 = (q & 127) * 4;
            float4 cv = *(const float4*)(centroids + (size_t)(cbase + br) * DIM + c4);
            st_bf16x4(&Blds[br * LDS_STRIDE + c4], cv.x, cv.y, cv.z, cv.w);
        }
        __syncthreads();

        v8f acc[NTILES];
        #pragma unroll
        for (int t = 0; t < NTILES; ++t)
            acc[t] = (v8f){0.f, 0.f, 0.f, 0.f, 0.f, 0.f, 0.f, 0.f};

        for (int kc = 0; kc < KCHUNKS; ++kc) {
            const int ko = kc * 32;
            v16bf af = ld_frag_a(&Alds[arow * LDS_STRIDE + ko + kb]);
            #pragma unroll
            for (int t = 0; t < NTILES; ++t) {
                v16bf bf = ld_frag_b(&Blds[(t * 16 + (lane & 15)) * LDS_STRIDE + ko + bko]);
                acc[t] = __builtin_amdgcn_wmma_f32_16x16x32_bf16(
                    false, af, false, bf, (short)0, acc[t], false, false);
            }
        }

        #pragma unroll
        for (int t = 0; t < NTILES; ++t) {
            const int n   = cbase + t * 16 + (lane & 15);
            const float c2v = c2[n];
            #pragma unroll
            for (int j = 0; j < 8; ++j) {
                const float d = c2v - 2.f * acc[t][j];   // |x|^2 term constant
                if (d < best_d[j]) { best_d[j] = d; best_i[j] = n; }
            }
        }
    }

    // ---- Phase 3: butterfly argmin within each 16-lane half ----
    #pragma unroll
    for (int j = 0; j < 8; ++j) {
        #pragma unroll
        for (int m = 8; m >= 1; m >>= 1) {
            const float od = __shfl_xor(best_d[j], m, 32);
            const int   oi = __shfl_xor(best_i[j], m, 32);
            if (od < best_d[j] || (od == best_d[j] && oi < best_i[j])) {
                best_d[j] = od; best_i[j] = oi;
            }
        }
    }

    // ---- Phase 4: outputs (index, quantized gather, counts/dw atomics) ----
    #pragma unroll
    for (int r = 0; r < 16; ++r) {
        const int idx = __shfl(best_i[r & 7], (r >> 3) * 16, 32);
        const size_t grow = rowBase + (size_t)w * ROWS_PER_WAVE + r;
        if (lane == 0) {
            out_idx[grow] = (float)idx;
            atomicAdd(&counts[idx], 1.0f);
        }
        const float* crow = centroids + (size_t)idx * DIM;
        const float* xrow = X + grow * DIM;
        float* qrow  = out_q + grow * DIM;
        float* dwrow = dw + (size_t)idx * DIM;
        #pragma unroll
        for (int i = 0; i < 4; ++i) {
            const int c = i * 128 + lane * 4;
            float4 cv = *(const float4*)(crow + c);
            *(float4*)(qrow + c) = cv;
            float4 xv = *(const float4*)(xrow + c);
            atomicAdd(dwrow + c + 0, xv.x);
            atomicAdd(dwrow + c + 1, xv.y);
            atomicAdd(dwrow + c + 2, xv.z);
            atomicAdd(dwrow + c + 3, xv.w);
        }
    }
}

// ---------------------------------------------------------------------------
__global__ void zero_ws_kernel(float* p, int n) {
    const int i = blockIdx.x * blockDim.x + threadIdx.x;
    if (i < n) p[i] = 0.f;
}

__global__ void prep_c2_kernel(const float* __restrict__ centroids,
                               float* __restrict__ c2) {
    const int row = blockIdx.x;
    const int tid = threadIdx.x;                 // 128 threads
    float4 v = *(const float4*)(centroids + (size_t)row * DIM + tid * 4);
    float ss = v.x * v.x + v.y * v.y + v.z * v.z + v.w * v.w;
    #pragma unroll
    for (int m = 16; m >= 1; m >>= 1) ss += __shfl_xor(ss, m, 32);
    __shared__ float red[4];
    if ((tid & 31) == 0) red[tid >> 5] = ss;
    __syncthreads();
    if (tid == 0) c2[row] = red[0] + red[1] + red[2] + red[3];
}

__global__ void finalize_size_kernel(const float* __restrict__ ema_size,
                                     const float* __restrict__ counts,
                                     float* __restrict__ out_nsize,
                                     float* __restrict__ ws_size) {
    __shared__ float red[NCLUST];
    const int k = threadIdx.x;                   // 512 threads, 1 block
    const float ns = ema_size[k] * DECAY + (1.f - DECAY) * counts[k];
    red[k] = ns;
    __syncthreads();
    for (int s = NCLUST / 2; s > 0; s >>= 1) {
        if (k < s) red[k] += red[k + s];
        __syncthreads();
    }
    const float n = red[0];
    const float v = (ns + EPS) / (n + (float)NCLUST * EPS) * n;
    out_nsize[k] = v;
    ws_size[k]   = v;
}

__global__ void finalize_w_kernel(const float* __restrict__ ema_w,
                                  const float* __restrict__ dw,
                                  const float* __restrict__ ws_size,
                                  float* __restrict__ out_emaw,
                                  float* __restrict__ out_cent) {
    const int i = blockIdx.x * blockDim.x + threadIdx.x;   // < 512*512
    const float v = ema_w[i] * DECAY + (1.f - DECAY) * dw[i];
    out_emaw[i] = v;
    out_cent[i] = v / ws_size[i >> 9];
}

// ---------------------------------------------------------------------------
extern "C" void kernel_launch(void* const* d_in, const int* in_sizes, int n_in,
                              void* d_out, int out_size, void* d_ws, size_t ws_size_bytes,
                              hipStream_t stream) {
    (void)in_sizes; (void)n_in; (void)out_size; (void)ws_size_bytes;

    const float* X         = (const float*)d_in[0];
    const float* centroids = (const float*)d_in[1];
    const float* ema_size  = (const float*)d_in[2];
    const float* ema_w     = (const float*)d_in[3];

    // output layout: quantized | indices | new_centroids | new_size | new_ema_w
    float* out       = (float*)d_out;
    float* out_q     = out;                                   // 131072*512
    float* out_idx   = out_q   + (size_t)NPOINTS * DIM;       // 131072
    float* out_cent  = out_idx + NPOINTS;                     // 512*512
    float* out_nsize = out_cent + (size_t)NCLUST * DIM;       // 512
    float* out_emaw  = out_nsize + NCLUST;                    // 512*512

    // workspace: counts | dw | ws_size | c2
    float* ws      = (float*)d_ws;
    float* counts  = ws;                                      // 512
    float* dw      = counts + NCLUST;                         // 512*512
    float* wssize  = dw + (size_t)NCLUST * DIM;               // 512
    float* c2      = wssize + NCLUST;                         // 512

    const int nzero = NCLUST + NCLUST * DIM;                  // counts + dw
    zero_ws_kernel<<<(nzero + 255) / 256, 256, 0, stream>>>(ws, nzero);
    prep_c2_kernel<<<NCLUST, 128, 0, stream>>>(centroids, c2);
    kmeans_assign_kernel<<<NPOINTS / ROWS_PER_WG, WG_THREADS, 0, stream>>>(
        X, centroids, c2, out_q, out_idx, counts, dw);
    finalize_size_kernel<<<1, NCLUST, 0, stream>>>(ema_size, counts, out_nsize, wssize);
    finalize_w_kernel<<<(NCLUST * DIM) / 256, 256, 0, stream>>>(
        ema_w, dw, wssize, out_emaw, out_cent);
}
